// MultiHeadAttentionQuantizer_11759620456774
// MI455X (gfx1250) — matrix operand order repaired and checked by reading, not compiled, not run on hardware
//
#include <hip/hip_runtime.h>
#include <cstdint>
#include <cstddef>

// Problem sizes (match reference)
constexpr int E_  = 1024;
constexpr int H_  = 16;
constexpr int DH_ = 64;
constexpr int K_  = 8;
constexpr int B_  = 4;
constexpr int S_  = 2048;
constexpr int N_  = B_ * S_;   // 8192 rows of representations
constexpr int KH_ = K_ * H_;   // 128 (k,h) score columns

typedef __attribute__((ext_vector_type(16))) __bf16 v16bf;
typedef __attribute__((ext_vector_type(8)))  float  v8f;

__device__ __forceinline__ unsigned short f2bf(float f) {
  union { float f; unsigned u; } v; v.f = f;
  unsigned r = v.u + 0x7fffu + ((v.u >> 16) & 1u);  // round-to-nearest-even
  return (unsigned short)(r >> 16);
}

__device__ __forceinline__ float warpRedSum(float v) {
#pragma unroll
  for (int o = 16; o > 0; o >>= 1) v += __shfl_down(v, o);
  return v;
}
__device__ __forceinline__ float warpRedMax(float v) {
#pragma unroll
  for (int o = 16; o > 0; o >>= 1) v = fmaxf(v, __shfl_down(v, o));
  return v;
}
__device__ __forceinline__ float blockRedSum(float v) {
  __shared__ float sh[32];
  int lane = threadIdx.x & 31, w = threadIdx.x >> 5, nw = blockDim.x >> 5;
  v = warpRedSum(v);
  if (lane == 0) sh[w] = v;
  __syncthreads();
  float r = (int(threadIdx.x) < nw) ? sh[threadIdx.x] : 0.0f;
  if (w == 0) { r = warpRedSum(r); if (lane == 0) sh[0] = r; }
  __syncthreads();
  r = sh[0];
  __syncthreads();
  return r;
}
__device__ __forceinline__ float blockRedMax(float v) {
  __shared__ float sh[32];
  int lane = threadIdx.x & 31, w = threadIdx.x >> 5, nw = blockDim.x >> 5;
  v = warpRedMax(v);
  if (lane == 0) sh[w] = v;
  __syncthreads();
  float r = (int(threadIdx.x) < nw) ? sh[threadIdx.x] : -3.4e38f;
  if (w == 0) { r = warpRedMax(r); if (lane == 0) sh[0] = r; }
  __syncthreads();
  r = sh[0];
  __syncthreads();
  return r;
}

// ---- Step 0: qs[k*E+f] = (queries[k,:]·Wq[k,f,:] + bq) * Dh^-0.5.  Streams Wq once.
__global__ void k_qproj(const float* __restrict__ queries, const float* __restrict__ Wq,
                        const float* __restrict__ bq, float* __restrict__ qs) {
  int lane = threadIdx.x & 31;
  int row  = blockIdx.x * (blockDim.x >> 5) + (threadIdx.x >> 5);  // 0..8191
  int k = row >> 10, f = row & 1023;
  const float* qr = queries + (size_t)k * E_;
  const float* wr = Wq + ((size_t)k * E_ + f) * E_;
  float acc = 0.f;
#pragma unroll 8
  for (int e = lane; e < E_; e += 32) acc += qr[e] * wr[e];
  acc = warpRedSum(acc);
  if (lane == 0) qs[row] = (acc + bq[row]) * 0.125f;  // 64^-0.5
}

// ---- Step 1: qk[kh,e] = sum_d qs[k, h*64+d] * Wk[k, h*64+d, e]  (bf16 out).  Streams Wk once.
__global__ void k_qk(const float* __restrict__ qs, const float* __restrict__ Wk,
                     unsigned short* __restrict__ qk) {
  int idx = blockIdx.x * blockDim.x + threadIdx.x;   // 0..131071
  int kh = idx >> 10, e = idx & 1023;
  int k = kh >> 4, h = kh & 15;
  const float* qsr = qs + (size_t)k * E_ + h * DH_;
  const float* wr  = Wk + ((size_t)k * E_ + h * DH_) * E_ + e;
  float acc = 0.f;
#pragma unroll 8
  for (int d = 0; d < DH_; ++d) acc += qsr[d] * wr[(size_t)d * E_];
  qk[idx] = f2bf(acc);
}

// ---- Step 2a: Xb = bf16(X), row-major [n][e]
__global__ void k_xb(const float* __restrict__ X, unsigned short* __restrict__ Xb) {
  int idx = blockIdx.x * blockDim.x + threadIdx.x;
  size_t base = (size_t)idx * 4;
  float4 v = *reinterpret_cast<const float4*>(X + base);
  union { unsigned short u[4]; uint2 p; } o;
  o.u[0] = f2bf(v.x); o.u[1] = f2bf(v.y); o.u[2] = f2bf(v.z); o.u[3] = f2bf(v.w);
  *reinterpret_cast<uint2*>(Xb + base) = o.p;
}

// ---- Step 2b: XbT[e][n] = bf16(X[n][e]).
// Stages the 32x32 fp32 tile through LDS with CDNA5 async global->LDS copies
// (global_load_async_to_lds_b128, tracked by ASYNCcnt), then reads transposed.
__global__ void k_xt(const float* __restrict__ X, unsigned short* __restrict__ XbT) {
  __shared__ float tileF[32][36];   // 144B row pitch: 16B aligned for B128 LDS writes
  const int e0 = blockIdx.x * 32, n0 = blockIdx.y * 32;
  const int tx = threadIdx.x, ty = threadIdx.y;          // block (32,8)
  const int tid = ty * 32 + tx;                          // 0..255
  const int r = tid >> 3;                                // tile row (n), 0..31
  const int c = (tid & 7) << 2;                          // tile col (e) start, 4 floats
  const float* gp = X + (size_t)(n0 + r) * E_ + (e0 + c);
  unsigned int ldsAddr = (unsigned int)(uintptr_t)&tileF[r][c];
  asm volatile("global_load_async_to_lds_b128 %0, %1, off"
               :: "v"(ldsAddr), "v"(gp) : "memory");
  asm volatile("s_wait_asynccnt 0" ::: "memory");
  __syncthreads();
#pragma unroll
  for (int j = 0; j < 32; j += 8)
    XbT[(size_t)(e0 + ty + j) * N_ + (n0 + tx)] = f2bf(tileF[tx][ty + j]);
}

// ---- Step 3: scores GEMM  P[(k,b,h)][s] = Xb[(b,s),:]·qk[kh,:]
// M=8192 (n), N=128 (kh), K=1024 (e).  Wave tile 16x64, bf16 WMMA, f32 acc.
// Ping-pong pipelined: operands for K-step i+1 are in flight while step i's WMMAs issue.
__global__ void k_scores(const unsigned short* __restrict__ Xb,
                         const unsigned short* __restrict__ qk,
                         float* __restrict__ P) {
  const int lane  = threadIdx.x & 31;
  const int wid   = blockIdx.x * (blockDim.x >> 5) + (threadIdx.x >> 5);  // 0..1023
  const int mT    = wid >> 1;
  const int nQ    = wid & 1;
  const int nBase = mT << 4;
  const int khBase = nQ << 6;
  const int laneL = lane & 15;
  const int hi    = lane >> 4;

  const unsigned short* aRow = Xb + (size_t)(nBase + laneL) * E_;
  const int aOff0 = hi ? 16 : 0;   // A frag: lo lanes K{0..15}, hi lanes K{16..31}
  const int aOff1 = aOff0 + 8;
  const unsigned short* bRow[4];
#pragma unroll
  for (int j = 0; j < 4; ++j)
    bRow[j] = qk + (size_t)(khBase + j * 16 + laneL) * E_ + (hi ? 16 : 0);

  auto LDA = [&](v16bf& a, int kk) {
    uint4* ap = reinterpret_cast<uint4*>(&a);
    ap[0] = *reinterpret_cast<const uint4*>(aRow + kk + aOff0);
    ap[1] = *reinterpret_cast<const uint4*>(aRow + kk + aOff1);
  };
  auto LDB = [&](v16bf* b, int kk) {
#pragma unroll
    for (int j = 0; j < 4; ++j) {
      uint4* bp = reinterpret_cast<uint4*>(&b[j]);
      const uint4* src = reinterpret_cast<const uint4*>(bRow[j] + kk);
      bp[0] = src[0]; bp[1] = src[1];
    }
  };

  v8f c[4] = {v8f{}, v8f{}, v8f{}, v8f{}};
  v16bf a0, a1, b0[4], b1[4];
  LDA(a0, 0); LDB(b0, 0);
  for (int kk = 0; kk < E_ - 64; kk += 64) {
    LDA(a1, kk + 32); LDB(b1, kk + 32);
#pragma unroll
    for (int j = 0; j < 4; ++j)
      c[j] = __builtin_amdgcn_wmma_f32_16x16x32_bf16(false, a0, false, b0[j],
                                                     (short)0, c[j], false, false);
    LDA(a0, kk + 64); LDB(b0, kk + 64);
#pragma unroll
    for (int j = 0; j < 4; ++j)
      c[j] = __builtin_amdgcn_wmma_f32_16x16x32_bf16(false, a1, false, b1[j],
                                                     (short)0, c[j], false, false);
  }
  LDA(a1, E_ - 32); LDB(b1, E_ - 32);
#pragma unroll
  for (int j = 0; j < 4; ++j)
    c[j] = __builtin_amdgcn_wmma_f32_16x16x32_bf16(false, a0, false, b0[j],
                                                   (short)0, c[j], false, false);
#pragma unroll
  for (int j = 0; j < 4; ++j)
    c[j] = __builtin_amdgcn_wmma_f32_16x16x32_bf16(false, a1, false, b1[j],
                                                   (short)0, c[j], false, false);

  // D layout: lane -> column (kh), VGPR r -> row M=r+8*hi.  Scatter into P[kbh][s].
  const int b_ = nBase >> 11;            // 16-row tile never crosses a batch boundary
#pragma unroll
  for (int j = 0; j < 4; ++j) {
    int kh = khBase + j * 16 + laneL;
    int k = kh >> 4, h = kh & 15;
    float* prow = P + ((size_t)((k * B_ + b_) * H_ + h) << 11);
#pragma unroll
    for (int r = 0; r < 8; ++r) {
      int s = (nBase + r + (hi << 3)) & (S_ - 1);
      prow[s] = c[j][r];
    }
  }
}

// ---- Step 4: softmax over s per (k,b,h) row, additive mask; bk drops (constant shift).
__global__ void k_softmax(const float* __restrict__ P, const float* __restrict__ mask,
                          unsigned short* __restrict__ Pb) {
  int kbh = blockIdx.x;
  int b = (kbh >> 4) & 3;
  const float* rowP = P + (size_t)kbh * S_;
  const float* mr   = mask + (size_t)b * S_;
  float m = -3.4e38f;
  for (int s = threadIdx.x; s < S_; s += blockDim.x) m = fmaxf(m, rowP[s] + mr[s]);
  m = blockRedMax(m);
  float sum = 0.f;
  for (int s = threadIdx.x; s < S_; s += blockDim.x) sum += __expf(rowP[s] + mr[s] - m);
  sum = blockRedSum(sum);
  float inv = 1.0f / sum;
  for (int s = threadIdx.x; s < S_; s += blockDim.x)
    Pb[(size_t)kbh * S_ + s] = f2bf(__expf(rowP[s] + mr[s] - m) * inv);
}

// ---- Step 5: xa GEMM  xa[(k,b,h)][e] = sum_s Pb[kbh][s] * Xb[(b,s)][e]
// M=16 (h, per (k,b) tile), N=1024 (e), K=2048 (s).  B from XbT -> contiguous loads.
__global__ void k_xa(const unsigned short* __restrict__ Pb,
                     const unsigned short* __restrict__ XbT,
                     float* __restrict__ xa) {
  const int lane = threadIdx.x & 31;
  const int wid  = blockIdx.x * (blockDim.x >> 5) + (threadIdx.x >> 5);  // 0..511
  const int kb = wid >> 4;
  const int nQ = wid & 15;
  const int b  = kb & 3;
  const int eBase = nQ << 6;
  const int laneL = lane & 15;
  const int hi    = lane >> 4;

  const unsigned short* aRow = Pb + (size_t)(kb * 16 + laneL) * S_;
  const int aOff0 = hi ? 16 : 0;
  const int aOff1 = aOff0 + 8;
  const unsigned short* bRow[4];
#pragma unroll
  for (int j = 0; j < 4; ++j)
    bRow[j] = XbT + (size_t)(eBase + j * 16 + laneL) * N_ + b * S_ + (hi ? 16 : 0);

  auto LDA = [&](v16bf& a, int kk) {
    uint4* ap = reinterpret_cast<uint4*>(&a);
    ap[0] = *reinterpret_cast<const uint4*>(aRow + kk + aOff0);
    ap[1] = *reinterpret_cast<const uint4*>(aRow + kk + aOff1);
  };
  auto LDB = [&](v16bf* bb, int kk) {
#pragma unroll
    for (int j = 0; j < 4; ++j) {
      uint4* bp = reinterpret_cast<uint4*>(&bb[j]);
      const uint4* src = reinterpret_cast<const uint4*>(bRow[j] + kk);
      bp[0] = src[0]; bp[1] = src[1];
    }
  };

  v8f c[4] = {v8f{}, v8f{}, v8f{}, v8f{}};
  v16bf a0, a1, b0[4], b1[4];
  LDA(a0, 0); LDB(b0, 0);
  for (int kk = 0; kk < S_ - 64; kk += 64) {
    LDA(a1, kk + 32); LDB(b1, kk + 32);
#pragma unroll
    for (int j = 0; j < 4; ++j)
      c[j] = __builtin_amdgcn_wmma_f32_16x16x32_bf16(false, a0, false, b0[j],
                                                     (short)0, c[j], false, false);
    LDA(a0, kk + 64); LDB(b0, kk + 64);
#pragma unroll
    for (int j = 0; j < 4; ++j)
      c[j] = __builtin_amdgcn_wmma_f32_16x16x32_bf16(false, a1, false, b1[j],
                                                     (short)0, c[j], false, false);
  }
  LDA(a1, S_ - 32); LDB(b1, S_ - 32);
#pragma unroll
  for (int j = 0; j < 4; ++j)
    c[j] = __builtin_amdgcn_wmma_f32_16x16x32_bf16(false, a0, false, b0[j],
                                                   (short)0, c[j], false, false);
#pragma unroll
  for (int j = 0; j < 4; ++j)
    c[j] = __builtin_amdgcn_wmma_f32_16x16x32_bf16(false, a1, false, b1[j],
                                                   (short)0, c[j], false, false);

#pragma unroll
  for (int j = 0; j < 4; ++j) {
    int e = eBase + j * 16 + laneL;
#pragma unroll
    for (int r = 0; r < 8; ++r) {
      int h = r + (hi << 3);
      xa[(size_t)(kb * 16 + h) * E_ + e] = c[j][r];
    }
  }
}

// ---- Step 6: ctx[k,b,f] = xa[(k,b,h=f>>6),:]·Wv[k,f,:] + bv[k,f].  Streams Wv once.
__global__ void k_ctx(const float* __restrict__ xa, const float* __restrict__ Wv,
                      const float* __restrict__ bv, float* __restrict__ ctx) {
  int lane = threadIdx.x & 31;
  int rid  = blockIdx.x * (blockDim.x >> 5) + (threadIdx.x >> 5);  // 0..32767
  int k = rid >> 12, b = (rid >> 10) & 3, f = rid & 1023, h = f >> 6;
  const float* xr = xa + (size_t)((k * B_ + b) * H_ + h) * E_;
  const float* wr = Wv + ((size_t)k * E_ + f) * E_;
  float acc = 0.f;
#pragma unroll 8
  for (int e = lane; e < E_; e += 32) acc += xr[e] * wr[e];
  acc = warpRedSum(acc);
  if (lane == 0) ctx[rid] = acc + bv[(size_t)k * E_ + f];
}

// ---- Steps 7a/7b: out[k,b,f] = in[k,b,:]·W[k,f,:] + bias[k,f].  Streams Wo / Wp once.
__global__ void k_proj(const float* __restrict__ in, const float* __restrict__ W,
                       const float* __restrict__ bias, float* __restrict__ out) {
  int lane = threadIdx.x & 31;
  int rid  = blockIdx.x * (blockDim.x >> 5) + (threadIdx.x >> 5);  // 0..32767
  int k = rid >> 12, b = (rid >> 10) & 3, f = rid & 1023;
  const float* ir = in + (size_t)(k * B_ + b) * E_;
  const float* wr = W + ((size_t)k * E_ + f) * E_;
  float acc = 0.f;
#pragma unroll 8
  for (int e = lane; e < E_; e += 32) acc += ir[e] * wr[e];
  acc = warpRedSum(acc);
  if (lane == 0) out[rid] = acc + bias[(size_t)k * E_ + f];
}

// ---- Step 8: LayerNorm over E, with (k,b) -> (b,k) transpose into d_out
__global__ void k_ln(const float* __restrict__ inT, const float* __restrict__ gamma,
                     const float* __restrict__ beta, float* __restrict__ out) {
  int b = blockIdx.x >> 3, kq = blockIdx.x & 7;
  const float* row = inT + (size_t)(kq * B_ + b) * E_;
  float s = 0.f, s2 = 0.f;
  for (int e = threadIdx.x; e < E_; e += blockDim.x) { float v = row[e]; s += v; s2 += v * v; }
  s = blockRedSum(s);
  s2 = blockRedSum(s2);
  float mu  = s * (1.0f / E_);
  float var = s2 * (1.0f / E_) - mu * mu;
  float inv = rsqrtf(var + 1e-5f);
  float* orow = out + (size_t)(b * K_ + kq) * E_;
  for (int e = threadIdx.x; e < E_; e += blockDim.x)
    orow[e] = gamma[e] * (row[e] - mu) * inv + beta[e];
}

// Workspace layout (bytes)
constexpr size_t OFF_QS   = 0;
constexpr size_t OFF_QK   = OFF_QS  + (size_t)K_ * E_ * 4;
constexpr size_t OFF_XB   = OFF_QK  + (size_t)KH_ * E_ * 2;
constexpr size_t OFF_XBT  = OFF_XB  + (size_t)N_ * E_ * 2;
constexpr size_t OFF_P    = OFF_XBT + (size_t)E_ * N_ * 2;
constexpr size_t OFF_PB   = OFF_P   + (size_t)KH_ * B_ * S_ * 4;
constexpr size_t OFF_XA   = OFF_PB  + (size_t)KH_ * B_ * S_ * 2;
constexpr size_t OFF_CTX  = OFF_XA  + (size_t)KH_ * B_ * E_ * 4 / H_;
constexpr size_t OFF_ATT  = OFF_CTX + (size_t)K_ * B_ * E_ * 4;
constexpr size_t OFF_OUTT = OFF_ATT + (size_t)K_ * B_ * E_ * 4;

extern "C" void kernel_launch(void* const* d_in, const int* in_sizes, int n_in,
                              void* d_out, int out_size, void* d_ws, size_t ws_size,
                              hipStream_t stream) {
  (void)in_sizes; (void)n_in; (void)out_size; (void)ws_size;
  const float* X       = (const float*)d_in[0];
  const float* mask    = (const float*)d_in[1];
  const float* queries = (const float*)d_in[2];
  const float* Wq = (const float*)d_in[3];  const float* bq = (const float*)d_in[4];
  const float* Wk = (const float*)d_in[5];  /* bk (d_in[6]) cancels in softmax */
  const float* Wv = (const float*)d_in[7];  const float* bv = (const float*)d_in[8];
  const float* Wo = (const float*)d_in[9];  const float* bo = (const float*)d_in[10];
  const float* Wp = (const float*)d_in[11]; const float* bp = (const float*)d_in[12];
  const float* gamma = (const float*)d_in[13];
  const float* beta  = (const float*)d_in[14];
  float* out = (float*)d_out;

  char* w = (char*)d_ws;
  float*          qs   = (float*)(w + OFF_QS);
  unsigned short* qk   = (unsigned short*)(w + OFF_QK);
  unsigned short* Xb   = (unsigned short*)(w + OFF_XB);
  unsigned short* XbT  = (unsigned short*)(w + OFF_XBT);
  float*          P    = (float*)(w + OFF_P);
  unsigned short* Pb   = (unsigned short*)(w + OFF_PB);
  float*          xa   = (float*)(w + OFF_XA);
  float*          ctx  = (float*)(w + OFF_CTX);
  float*          attn = (float*)(w + OFF_ATT);
  float*          outT = (float*)(w + OFF_OUTT);

  k_qproj  <<<1024, 256, 0, stream>>>(queries, Wq, bq, qs);
  k_qk     <<<512, 256, 0, stream>>>(qs, Wk, qk);
  k_xb     <<<8192, 256, 0, stream>>>(X, Xb);
  k_xt     <<<dim3(E_ / 32, N_ / 32), dim3(32, 8), 0, stream>>>(X, XbT);
  k_scores <<<256, 128, 0, stream>>>(Xb, qk, P);
  k_softmax<<<KH_ * B_, 256, 0, stream>>>(P, mask, Pb);
  k_xa     <<<128, 128, 0, stream>>>(Pb, XbT, xa);
  k_ctx    <<<4096, 256, 0, stream>>>(xa, Wv, bv, ctx);
  k_proj   <<<4096, 256, 0, stream>>>(ctx, Wo, bo, attn);
  k_proj   <<<4096, 256, 0, stream>>>(attn, Wp, bp, outT);
  k_ln     <<<B_ * K_, 256, 0, stream>>>(outT, gamma, beta, out);
}
